// GridHead_3564822856338
// MI455X (gfx1250) — compile-verified, matrix operand order checked by reading, not run on hardware
//
#include <hip/hip_runtime.h>
#include <hip/hip_fp16.h>

typedef _Float16 half16 __attribute__((ext_vector_type(16)));
typedef float    float8 __attribute__((ext_vector_type(8)));

// ---------------------------------------------------------------------------
// Hardcoded 3x3 grid edge structure (matches reference _edges() order).
// ---------------------------------------------------------------------------
__constant__ int c_es[9]   = {0, 2, 5, 7, 10, 14, 17, 19, 22};
__constant__ int c_ec[9]   = {2, 3, 2, 3, 4, 3, 2, 3, 2};
__constant__ int c_src[24] = {1,3, 0,2,4, 1,5, 0,4,6, 1,3,5,7, 2,4,8, 3,7, 4,6,8, 5,7};

// ---------------------------------------------------------------------------
// Layouts:
//   activations: halo-padded NHWC f16, pad=1 on each spatial side:
//     act[((n*(H+2) + h+1)*(W+2) + (w+1))*C + c], borders zeroed once.
//     -> every 3x3 im2col tap address is valid; no bounds checks in the GEMM.
//   conv GEMM out: dense f32, m-major (m = (n*Hout+oh)*Wout+ow)
//   fragment-packed weights (WMMA B-operand ready):
//     wfrag[((chunk*Cout + col)*2 + h)*16 + j]
//       chunk = K/32 chunk (K = r*Cin + ic, r = kh*3+kw)
//       h     = 0 for lanes 0-15, 1 for lanes 16-31
//       j<8   -> B[chunk*32 + h*8 + j][col]
//       j>=8  -> B[chunk*32 + 16 + h*8 + (j-8)][col]
// ---------------------------------------------------------------------------

// padded-activation index helper for the 7x7/C=576 stage buffers (Hp=Wp=9)
__device__ __forceinline__ size_t act_idx(int n, int hw) {
  int h = hw / 7, w = hw - h * 7;
  return ((size_t)(n * 9 + h + 1) * 9 + (w + 1)) * 576;
}

__global__ __launch_bounds__(256)
void zero_u16x8(uint4* __restrict__ dst, size_t count16) {
  size_t t = (size_t)blockIdx.x * 256 + threadIdx.x;
  if (t < count16) dst[t] = make_uint4(0u, 0u, 0u, 0u);
}

// input NCHW f32 -> halo-padded NHWC f16 (14x14 -> 16x16, C=256)
__global__ __launch_bounds__(256)
void nchw_to_nhwc_pad_f16(const float* __restrict__ src, _Float16* __restrict__ dst) {
  size_t t = (size_t)blockIdx.x * 256 + threadIdx.x;   // over 512*256*196
  size_t n = t / (256 * 196);
  size_t rem = t - n * (256 * 196);
  size_t c = rem / 196;
  size_t hw = rem - c * 196;
  size_t h = hw / 14, w = hw - h * 14;
  dst[((n * 16 + h + 1) * 16 + (w + 1)) * 256 + c] = (_Float16)src[t];
}

// Pack conv weights [Cout][Cin][3][3] f32 -> fragment-ready f16 (layout above).
__global__ __launch_bounds__(256)
void pack_w_frag(const float* __restrict__ src, _Float16* __restrict__ dst,
                 int Cout, int Cin) {
  int t = blockIdx.x * 256 + threadIdx.x;
  int total = Cout * Cin * 9;
  if (t >= total) return;
  int oc = t / (Cin * 9);
  int rem = t - oc * (Cin * 9);
  int ic = rem / 9;
  int r  = rem - ic * 9;
  int k  = r * Cin + ic;
  int c  = k >> 5;
  int kk = k & 31;
  int h  = (kk >> 3) & 1;
  int j  = (kk & 7) + ((kk & 16) ? 8 : 0);
  dst[((size_t)(c * Cout + oc) * 2 + h) * 16 + j] = (_Float16)src[t];
}

// ---------------------------------------------------------------------------
// Implicit-GEMM 3x3 conv (pad via memory halo, stride 1/2) using
// v_wmma_f32_16x16x32_f16. No LDS, no bounds checks: A fragments are two
// unconditional b128 loads from the padded NHWC activations; B fragments are
// two b128 loads from fragment-packed weights. All operand addresses are
// constant-stride induction variables.
// Block = 256 threads = 8 waves (4Mx2N); wave tile 32x32 (4 accumulators);
// block tile 128(M) x 64(N). Grid 196 x 9 (exact, no partial tiles).
// ---------------------------------------------------------------------------
template<int CIN>
__global__ __launch_bounds__(256)
void conv3x3_wmma(const _Float16* __restrict__ act, const _Float16* __restrict__ wfrag,
                  float* __restrict__ out, int Hp, int Wp,
                  int Hout, int Wout, int Cout, int stride) {
  constexpr int CHUNKS_R = CIN / 32;   // K-chunks per (kh,kw) tap

  const int t    = threadIdx.x;
  const int lane = t & 31;
  const int wv   = t >> 5;
  const int wm   = wv >> 1;            // 0..3
  const int wn   = wv & 1;             // 0..1

  const int mBase  = blockIdx.x * 128 + wm * 32;
  const int ocBase = blockIdx.y * 64 + wn * 32;

  const int lrow = lane & 15;
  const int kh8  = (lane < 16) ? 0 : 8;   // A half-select
  const int hB   = (lane < 16) ? 0 : 1;   // B half-select

  // two A rows per lane (wave tile has 32 rows)
  const int HWo = Hout * Wout;
  const int m0 = mBase + lrow;
  const int m1 = m0 + 16;
  const int n0 = m0 / HWo, rm0 = m0 - n0 * HWo;
  const int oh0 = rm0 / Wout, ow0 = rm0 - oh0 * Wout;
  const int n1 = m1 / HWo, rm1 = m1 - n1 * HWo;
  const int oh1 = rm1 / Wout, ow1 = rm1 - oh1 * Wout;

  // B pointers: uniform induction variables over the whole flattened K loop
  const size_t bstep = (size_t)Cout * 4;   // uint4 per 32-K chunk
  const uint4* bq0 = reinterpret_cast<const uint4*>(wfrag)
                   + ((size_t)(ocBase + lrow) * 2 + hB) * 2;
  const uint4* bq1 = bq0 + 32 * 2;         // +16 cols

  float8 c00 = {}, c01 = {}, c10 = {}, c11 = {};

  for (int r = 0; r < 9; ++r) {
    const int kh = r / 3, kw = r - kh * 3;
    // halo padding: tap (kh,kw) maps to padded coords without -1; always valid
    const uint4* ap0 = reinterpret_cast<const uint4*>(
        act + ((size_t)(n0 * Hp + oh0 * stride + kh) * Wp + ow0 * stride + kw) * CIN + kh8);
    const uint4* ap1 = reinterpret_cast<const uint4*>(
        act + ((size_t)(n1 * Hp + oh1 * stride + kh) * Wp + ow1 * stride + kw) * CIN + kh8);

#pragma unroll 2
    for (int cb = 0; cb < CHUNKS_R; ++cb) {
      union { uint4 u[2]; half16 h; } A0, A1, B0, B1;
      A0.u[0] = ap0[0]; A0.u[1] = ap0[2];
      A1.u[0] = ap1[0]; A1.u[1] = ap1[2];
      B0.u[0] = bq0[0]; B0.u[1] = bq0[1];
      B1.u[0] = bq1[0]; B1.u[1] = bq1[1];

      ap0 += 4; ap1 += 4;                  // next 32 input channels
      bq0 += bstep; bq1 += bstep;          // next K chunk

      c00 = __builtin_amdgcn_wmma_f32_16x16x32_f16(false, A0.h, false, B0.h, (short)0, c00, false, false);
      c01 = __builtin_amdgcn_wmma_f32_16x16x32_f16(false, A0.h, false, B1.h, (short)0, c01, false, false);
      c10 = __builtin_amdgcn_wmma_f32_16x16x32_f16(false, A1.h, false, B0.h, (short)0, c10, false, false);
      c11 = __builtin_amdgcn_wmma_f32_16x16x32_f16(false, A1.h, false, B1.h, (short)0, c11, false, false);
    }
  }

  // epilogue: f32 C/D layout (lane<16 -> M=rr, lane>=16 -> M=rr+8; N=lane&15)
  const int rofs = (lane < 16) ? 0 : 8;
#pragma unroll
  for (int rr = 0; rr < 8; ++rr) {
    const int mg0 = mBase + rofs + rr;
    const int mg1 = mg0 + 16;
    const size_t o0 = (size_t)mg0 * Cout + ocBase;
    const size_t o1 = (size_t)mg1 * Cout + ocBase;
    out[o0 + lrow]      = c00[rr];
    out[o0 + 16 + lrow] = c01[rr];
    out[o1 + lrow]      = c10[rr];
    out[o1 + 16 + lrow] = c11[rr];
  }
}

// ---------------------------------------------------------------------------
// GroupNorm(36 groups of 16 ch) + conv-bias + ReLU.
// In: dense f32 m-major GEMM output (512*49 x 576).
// Out: halo-padded NHWC f16 stage buffer (borders stay zero).
// One block per (n, group).
// ---------------------------------------------------------------------------
__global__ __launch_bounds__(256)
void gn_relu_kernel(const float* __restrict__ in, const float* __restrict__ bias,
                    const float* __restrict__ gamma, const float* __restrict__ beta,
                    _Float16* __restrict__ out) {
  __shared__ float rs[256];
  __shared__ float rq[256];
  const int nIdx = blockIdx.x / 36;
  const int g    = blockIdx.x - nIdx * 36;

  float s = 0.f, q = 0.f;
  for (int e = threadIdx.x; e < 784; e += 256) {
    int hw = e >> 4;
    int c  = (g << 4) + (e & 15);
    float v = in[((size_t)nIdx * 49 + hw) * 576 + c] + bias[c];
    s += v; q += v * v;
  }
  rs[threadIdx.x] = s; rq[threadIdx.x] = q;
  __syncthreads();
  for (int off = 128; off > 0; off >>= 1) {
    if (threadIdx.x < off) {
      rs[threadIdx.x] += rs[threadIdx.x + off];
      rq[threadIdx.x] += rq[threadIdx.x + off];
    }
    __syncthreads();
  }
  const float mean = rs[0] / 784.f;
  const float var  = rq[0] / 784.f - mean * mean;
  const float sc   = rsqrtf(var + 1e-5f);

  for (int e = threadIdx.x; e < 784; e += 256) {
    int hw = e >> 4;
    int c  = (g << 4) + (e & 15);
    float v = in[((size_t)nIdx * 49 + hw) * 576 + c] + bias[c];
    float y = (v - mean) * sc * gamma[c] + beta[c];
    out[act_idx(nIdx, hw) + c] = (_Float16)fmaxf(y, 0.f);
  }
}

// ---------------------------------------------------------------------------
// Grid-point message passing: out[n,p] = base[n,p] + sum_{e: dst(e)=p}
//   pw_e @ (dw5x5_e * srcfeat[n, src(e)]) + biases.  One block per (n,p).
// All activation buffers use the padded layout (interior only touched).
// ---------------------------------------------------------------------------
__global__ __launch_bounds__(256)
void msgpass_kernel(const _Float16* __restrict__ base, const _Float16* __restrict__ srcf,
                    const float* __restrict__ dw_w, const float* __restrict__ dw_b,
                    const float* __restrict__ pw_w, const float* __restrict__ pw_b,
                    _Float16* __restrict__ out) {
  __shared__ float acc[64 * 49];
  __shared__ float sf[64 * 49];
  __shared__ float yb[64 * 49];
  const int n = blockIdx.x / 9;
  const int p = blockIdx.x - n * 9;

  for (int idx = threadIdx.x; idx < 64 * 49; idx += 256) {
    int cc = idx / 49, hw = idx - cc * 49;
    acc[idx] = (float)base[act_idx(n, hw) + p * 64 + cc];
  }
  const int e0 = c_es[p], cnt = c_ec[p];
  for (int ei = 0; ei < cnt; ++ei) {
    const int e = e0 + ei;
    const int qsrc = c_src[e];
    for (int idx = threadIdx.x; idx < 64 * 49; idx += 256) {
      int cc = idx / 49, hw = idx - cc * 49;
      sf[idx] = (float)srcf[act_idx(n, hw) + qsrc * 64 + cc];
    }
    __syncthreads();
    // depthwise 5x5, pad 2
    for (int idx = threadIdx.x; idx < 64 * 49; idx += 256) {
      int cc = idx / 49, hw = idx - cc * 49;
      int h = hw / 7, w = hw - h * 7;
      float s = dw_b[e * 64 + cc];
      const float* dwp = dw_w + (size_t)(e * 64 + cc) * 25;
#pragma unroll
      for (int kh = 0; kh < 5; ++kh) {
        int hh = h + kh - 2;
        if (hh < 0 || hh >= 7) continue;
#pragma unroll
        for (int kw = 0; kw < 5; ++kw) {
          int ww = w + kw - 2;
          if (ww < 0 || ww >= 7) continue;
          s += dwp[kh * 5 + kw] * sf[cc * 49 + hh * 7 + ww];
        }
      }
      yb[idx] = s;
    }
    __syncthreads();
    // pointwise 64x64
    for (int idx = threadIdx.x; idx < 64 * 49; idx += 256) {
      int o = idx / 49, hw = idx - o * 49;
      float s = pw_b[e * 64 + o];
      const float* wr = pw_w + (size_t)(e * 64 + o) * 64;
#pragma unroll 8
      for (int c2 = 0; c2 < 64; ++c2) s += wr[c2] * yb[c2 * 49 + hw];
      acc[idx] += s;
    }
    __syncthreads();
  }
  for (int idx = threadIdx.x; idx < 64 * 49; idx += 256) {
    int cc = idx / 49, hw = idx - cc * 49;
    out[act_idx(n, hw) + p * 64 + cc] = (_Float16)acc[idx];
  }
}

// ---------------------------------------------------------------------------
// Fully fused decoder per (n,p): grouped ConvT4x4 s2 (64->64) -> GN(P)+ReLU
// -> grouped ConvT4x4 s2 (64->1) -> d_out. GN group == block's channel tile.
// ---------------------------------------------------------------------------
__global__ __launch_bounds__(256)
void decoder_kernel(const _Float16* __restrict__ act,
                    const float* __restrict__ w1, const float* __restrict__ b1v,
                    const float* __restrict__ g1, const float* __restrict__ be1,
                    const float* __restrict__ w2, const float* __restrict__ b2,
                    float* __restrict__ out) {
  __shared__ float fin[64 * 49];
  __shared__ float y1[64 * 196];
  __shared__ float rs[256];
  __shared__ float rq[256];
  const int n = blockIdx.x / 9;
  const int p = blockIdx.x - n * 9;

  for (int idx = threadIdx.x; idx < 64 * 49; idx += 256) {
    int ic = idx / 49, hw = idx - ic * 49;
    fin[ic * 49 + hw] = (float)act[act_idx(n, hw) + p * 64 + ic];
  }
  __syncthreads();

  // deconv1: oh = 2*ih - 1 + kh  ->  ih = (oh+1-kh)/2
  for (int idx = threadIdx.x; idx < 64 * 196; idx += 256) {
    int oc = idx / 196;
    int rmm = idx - oc * 196;
    int oh = rmm / 14, ow = rmm - oh * 14;
    float s = b1v[p * 64 + oc];
#pragma unroll
    for (int kh = 0; kh < 4; ++kh) {
      int th = oh + 1 - kh;
      if (th < 0 || (th & 1)) continue;
      int ih = th >> 1; if (ih >= 7) continue;
#pragma unroll
      for (int kw = 0; kw < 4; ++kw) {
        int tw = ow + 1 - kw;
        if (tw < 0 || (tw & 1)) continue;
        int iw = tw >> 1; if (iw >= 7) continue;
        const float* wpp = w1 + ((size_t)(p * 64) * 64 + oc) * 16 + kh * 4 + kw;
        float ss = 0.f;
        for (int ic = 0; ic < 64; ++ic)
          ss += wpp[(size_t)ic * 1024] * fin[ic * 49 + ih * 7 + iw];
        s += ss;
      }
    }
    y1[idx] = s;
  }
  __syncthreads();

  // GN over this group's 64 x 14 x 14 tile
  float s = 0.f, q = 0.f;
  for (int idx = threadIdx.x; idx < 64 * 196; idx += 256) {
    float v = y1[idx]; s += v; q += v * v;
  }
  rs[threadIdx.x] = s; rq[threadIdx.x] = q;
  __syncthreads();
  for (int off = 128; off > 0; off >>= 1) {
    if (threadIdx.x < off) {
      rs[threadIdx.x] += rs[threadIdx.x + off];
      rq[threadIdx.x] += rq[threadIdx.x + off];
    }
    __syncthreads();
  }
  const float mean = rs[0] / 12544.f;
  const float var  = rq[0] / 12544.f - mean * mean;
  const float sc   = rsqrtf(var + 1e-5f);
  for (int idx = threadIdx.x; idx < 64 * 196; idx += 256) {
    int oc = idx / 196;
    float v = (y1[idx] - mean) * sc * g1[p * 64 + oc] + be1[p * 64 + oc];
    y1[idx] = fmaxf(v, 0.f);
  }
  __syncthreads();

  // deconv2 (64 -> 1), 28x28 output
  for (int idx = threadIdx.x; idx < 784; idx += 256) {
    int oh = idx / 28, ow = idx - oh * 28;
    float s2 = b2[p];
#pragma unroll
    for (int kh = 0; kh < 4; ++kh) {
      int th = oh + 1 - kh;
      if (th < 0 || (th & 1)) continue;
      int ih = th >> 1; if (ih >= 14) continue;
#pragma unroll
      for (int kw = 0; kw < 4; ++kw) {
        int tw = ow + 1 - kw;
        if (tw < 0 || (tw & 1)) continue;
        int iw = tw >> 1; if (iw >= 14) continue;
        float ss = 0.f;
        for (int oc = 0; oc < 64; ++oc)
          ss += w2[(size_t)(p * 64 + oc) * 16 + kh * 4 + kw] * y1[oc * 196 + ih * 14 + iw];
        s2 += ss;
      }
    }
    out[((size_t)(n * 9 + p)) * 784 + idx] = s2;
  }
}

// ---------------------------------------------------------------------------
// Host side
// ---------------------------------------------------------------------------
extern "C" void kernel_launch(void* const* d_in, const int* in_sizes, int n_in,
                              void* d_out, int out_size, void* d_ws, size_t ws_size,
                              hipStream_t stream) {
  (void)in_sizes; (void)n_in; (void)out_size; (void)ws_size;

  const float* x        = (const float*)d_in[0];
  const float* conv0_w  = (const float*)d_in[1];
  const float* conv0_b  = (const float*)d_in[2];
  const float* convs_w  = (const float*)d_in[3];
  const float* convs_b  = (const float*)d_in[4];
  const float* gn_gamma = (const float*)d_in[5];
  const float* gn_beta  = (const float*)d_in[6];
  const float* fo_dw_w  = (const float*)d_in[7];
  const float* fo_dw_b  = (const float*)d_in[8];
  const float* fo_pw_w  = (const float*)d_in[9];
  const float* fo_pw_b  = (const float*)d_in[10];
  const float* so_dw_w  = (const float*)d_in[11];
  const float* so_dw_b  = (const float*)d_in[12];
  const float* so_pw_w  = (const float*)d_in[13];
  const float* so_pw_b  = (const float*)d_in[14];
  const float* dc1_w    = (const float*)d_in[15];
  const float* dc1_b    = (const float*)d_in[16];
  const float* gn1_g    = (const float*)d_in[17];
  const float* gn1_b    = (const float*)d_in[18];
  const float* dc2_w    = (const float*)d_in[19];
  const float* dc2_b    = (const float*)d_in[20];
  float* out = (float*)d_out;

  const int N = 512;
  const size_t XIN_E  = (size_t)N * 16 * 16 * 256;  // padded input, 33,554,432 halfs
  const size_t ACTP_E = (size_t)N * 9 * 9 * 576;    // padded stage,  23,887,872 halfs
  const size_t GEMM_E = (size_t)N * 49 * 576;       // dense GEMM out (f32)

  char* ws = (char*)d_ws;
  size_t off = 0;
  auto alloc = [&](size_t bytes) -> char* {
    char* p = ws + off;
    off = (off + bytes + 255) & ~(size_t)255;
    return p;
  };
  _Float16* xin   = (_Float16*)alloc(XIN_E * 2);
  _Float16* wp0   = (_Float16*)alloc((size_t)9 * 256 * 576 * 2);
  _Float16* wpL   = (_Float16*)alloc((size_t)9 * 576 * 576 * 2);
  float*    gemm  = (float*)   alloc(GEMM_E * 4);
  _Float16* actA  = (_Float16*)alloc(ACTP_E * 2);
  _Float16* actB  = (_Float16*)alloc(ACTP_E * 2);
  _Float16* fo_bf = (_Float16*)alloc(ACTP_E * 2);
  _Float16* so_bf = (_Float16*)alloc(ACTP_E * 2);

  // zero the halo-padded conv-input buffers (borders must be 0 every call)
  zero_u16x8<<<(unsigned)((XIN_E * 2 / 16 + 255) / 256), 256, 0, stream>>>(
      (uint4*)xin, XIN_E * 2 / 16);
  zero_u16x8<<<(unsigned)((ACTP_E * 2 / 16 + 255) / 256), 256, 0, stream>>>(
      (uint4*)actA, ACTP_E * 2 / 16);
  zero_u16x8<<<(unsigned)((ACTP_E * 2 / 16 + 255) / 256), 256, 0, stream>>>(
      (uint4*)actB, ACTP_E * 2 / 16);

  // input NCHW f32 -> padded NHWC f16
  nchw_to_nhwc_pad_f16<<<(unsigned)(((size_t)N * 256 * 196 + 255) / 256), 256, 0,
                         stream>>>(x, xin);

  // conv0: 256 -> 576, stride 2, padded 16x16 -> 7x7
  pack_w_frag<<<5184, 256, 0, stream>>>(conv0_w, wp0, 576, 256);
  conv3x3_wmma<256><<<dim3(196, 9), 256, 0, stream>>>(xin, wp0, gemm,
                                                      16, 16, 7, 7, 576, 2);
  gn_relu_kernel<<<N * 36, 256, 0, stream>>>(gemm, conv0_b, gn_gamma, gn_beta, actA);

  // 7 x [conv3x3 576->576 + GN + ReLU], padded 9x9 -> 7x7
  _Float16* cur = actA;
  _Float16* nxt = actB;
  for (int i = 0; i < 7; ++i) {
    pack_w_frag<<<11664, 256, 0, stream>>>(convs_w + (size_t)i * 576 * 576 * 9,
                                           wpL, 576, 576);
    conv3x3_wmma<576><<<dim3(196, 9), 256, 0, stream>>>(cur, wpL, gemm,
                                                        9, 9, 7, 7, 576, 1);
    gn_relu_kernel<<<N * 36, 256, 0, stream>>>(gemm, convs_b + (size_t)i * 576,
                                               gn_gamma + (size_t)(i + 1) * 576,
                                               gn_beta + (size_t)(i + 1) * 576, nxt);
    _Float16* tmp = cur; cur = nxt; nxt = tmp;
  }
  // cur = conv-stack output x (padded f16 NHWC)

  // grid-point message passing: first order, then second order
  msgpass_kernel<<<N * 9, 256, 0, stream>>>(cur, cur, fo_dw_w, fo_dw_b,
                                            fo_pw_w, fo_pw_b, fo_bf);
  msgpass_kernel<<<N * 9, 256, 0, stream>>>(cur, fo_bf, so_dw_w, so_dw_b,
                                            so_pw_w, so_pw_b, so_bf);

  // decoders: fused (x_so) then unfused (x)
  const size_t BRANCH = (size_t)N * 9 * 784;
  decoder_kernel<<<N * 9, 256, 0, stream>>>(so_bf, dc1_w, dc1_b, gn1_g, gn1_b,
                                            dc2_w, dc2_b, out);
  decoder_kernel<<<N * 9, 256, 0, stream>>>(cur, dc1_w, dc1_b, gn1_g, gn1_b,
                                            dc2_w, dc2_b, out + BRANCH);
}
